// MixtralMoE_67293547594307
// MI455X (gfx1250) — matrix operand order repaired and checked by reading, not compiled, not run on hardware
//
#include <hip/hip_runtime.h>
#include <hip/hip_bf16.h>
#include <stdint.h>

typedef __attribute__((ext_vector_type(16))) __bf16 v16bf;
typedef __attribute__((ext_vector_type(8)))  float  v8f;

#define T_TOK 2048
#define DDIM  2048
#define FDIM  7168
#define NEXP  8

union Frag { v16bf v; uint4 q[2]; };

__device__ __forceinline__ unsigned short f2bf(float f) {
    union { float f; unsigned int u; } v; v.f = f;
    unsigned int u = v.u;
    unsigned int r = u + 0x7FFFu + ((u >> 16) & 1u);   // round-to-nearest-even
    return (unsigned short)(r >> 16);
}
// single-instruction packed fp32->bf16 (RNE): v_cvt_pk_bf16_f32
__device__ __forceinline__ unsigned int pack2bf(float a, float b) {
    unsigned int r;
    asm("v_cvt_pk_bf16_f32 %0, %1, %2" : "=v"(r) : "v"(a), "v"(b));
    return r;
}

// ---------------------------------------------------------------- zero init
__global__ void k_zero(float* __restrict__ out, int n, int* __restrict__ cnt) {
    int i = blockIdx.x * 256 + threadIdx.x;
    if (i < n) out[i] = 0.0f;
    if (i < NEXP) cnt[i] = 0;
}

// ---------------------------------------------------------------- gating
__global__ void __launch_bounds__(256)
k_gate(const float* __restrict__ x, const float* __restrict__ gw,
       float* __restrict__ wslot, int* __restrict__ cnt, int* __restrict__ list) {
    const int t = blockIdx.x;
    __shared__ float red[256][NEXP];
    float acc[NEXP];
#pragma unroll
    for (int e = 0; e < NEXP; ++e) acc[e] = 0.0f;
    const float* xr = x + (size_t)t * DDIM;
    for (int d = threadIdx.x; d < DDIM; d += 256) {
        float xv = xr[d];
#pragma unroll
        for (int e = 0; e < NEXP; ++e) acc[e] += xv * gw[e * DDIM + d];
    }
#pragma unroll
    for (int e = 0; e < NEXP; ++e) red[threadIdx.x][e] = acc[e];
    __syncthreads();
    for (int s = 128; s > 0; s >>= 1) {
        if (threadIdx.x < (unsigned)s)
#pragma unroll
            for (int e = 0; e < NEXP; ++e) red[threadIdx.x][e] += red[threadIdx.x + s][e];
        __syncthreads();
    }
    if (threadIdx.x == 0) {
        float l[NEXP], p[NEXP], m = -1e30f, s = 0.0f;
#pragma unroll
        for (int e = 0; e < NEXP; ++e) { l[e] = red[0][e]; m = fmaxf(m, l[e]); }
#pragma unroll
        for (int e = 0; e < NEXP; ++e) { p[e] = __expf(l[e] - m); s += p[e]; }
        int e0 = 0;
#pragma unroll
        for (int e = 1; e < NEXP; ++e) if (p[e] > p[e0]) e0 = e;
        int e1 = (e0 == 0) ? 1 : 0;
#pragma unroll
        for (int e = 0; e < NEXP; ++e) if (e != e0 && p[e] > p[e1]) e1 = e;
        float ps = p[e0] + p[e1];
        float inv = __builtin_amdgcn_rcpf(ps);
        wslot[2 * t + 0] = p[e0] * inv;
        wslot[2 * t + 1] = p[e1] * inv;
        int pos0 = atomicAdd(&cnt[e0], 1); list[e0 * T_TOK + pos0] = 2 * t + 0;
        int pos1 = atomicAdd(&cnt[e1], 1); list[e1 * T_TOK + pos1] = 2 * t + 1;
    }
}

// ------------------------------------------------ GEMM1+GEMM3 fused SwiGLU
// Workgroup: one expert, 128 token rows x 128 F cols; 8 wave32s, wave owns
// 16 F cols, sweeps 8 row-subtiles per K-chunk. Software-pipelined: next
// K-chunk global loads are issued before the WMMA sweep of the current one.
__global__ void __launch_bounds__(256)
k_expert_gemm1(const float* __restrict__ x, const float* __restrict__ w1,
               const float* __restrict__ w3, const int* __restrict__ cnt,
               const int* __restrict__ list, unsigned short* __restrict__ act) {
    const int e = blockIdx.y >> 4;             // 16 row blocks per expert
    const int rowBlock = blockIdx.y & 15;
    const int n_e = cnt[e];
    if (rowBlock * 128 >= n_e) return;
    const int fTile = blockIdx.x * 128;
    const int tid = threadIdx.x;
    const int wave = tid >> 5;
    const int lane = tid & 31;
    const int ln = lane & 15, half = lane >> 4;

    __shared__ unsigned int xs[128][16];        // 128 tok x 32 d bf16 pairs
    __shared__ unsigned int w1s[8][16][16];     // per-wave 16 f x 32 d
    __shared__ unsigned int w3s[8][16][16];
    __shared__ int slots[128];

    if (tid < 128) {
        int r = rowBlock * 128 + tid;
        slots[tid] = (r < n_e) ? list[e * T_TOK + r] : -1;
    }
    __syncthreads();

    const float* w1e = w1 + (size_t)e * FDIM * DDIM;
    const float* w3e = w3 + (size_t)e * FDIM * DDIM;
    const int fbase = fTile + wave * 16;

    float4 xreg[4], areg[4], breg[4];

    v8f acch[8], accg[8];
#pragma unroll
    for (int s = 0; s < 8; ++s) { acch[s] = {}; accg[s] = {}; }

    // prologue: issue loads for chunk 0
#pragma unroll
    for (int q = 0; q < 4; ++q) {
        int idx = q * 256 + tid;
        int r = idx >> 3, c4 = idx & 7;
        int s = slots[r];
        xreg[q] = make_float4(0.f, 0.f, 0.f, 0.f);
        if (s >= 0) xreg[q] = *(const float4*)(x + (size_t)(s >> 1) * DDIM + c4 * 4);
    }
#pragma unroll
    for (int q = 0; q < 4; ++q) {
        int idx = q * 32 + lane;
        int r = idx >> 3, c4 = idx & 7;
        areg[q] = *(const float4*)(w1e + (size_t)(fbase + r) * DDIM + c4 * 4);
        breg[q] = *(const float4*)(w3e + (size_t)(fbase + r) * DDIM + c4 * 4);
    }

    for (int dk = 0; dk < DDIM; dk += 32) {
        __syncthreads();   // previous chunk's fragment reads complete
        // convert + store the staged chunk
#pragma unroll
        for (int q = 0; q < 4; ++q) {
            int idx = q * 256 + tid;
            int r = idx >> 3, c4 = idx & 7;
            xs[r][c4 * 2 + 0] = pack2bf(xreg[q].x, xreg[q].y);
            xs[r][c4 * 2 + 1] = pack2bf(xreg[q].z, xreg[q].w);
        }
#pragma unroll
        for (int q = 0; q < 4; ++q) {
            int idx = q * 32 + lane;
            int r = idx >> 3, c4 = idx & 7;
            w1s[wave][r][c4 * 2 + 0] = pack2bf(areg[q].x, areg[q].y);
            w1s[wave][r][c4 * 2 + 1] = pack2bf(areg[q].z, areg[q].w);
            w3s[wave][r][c4 * 2 + 0] = pack2bf(breg[q].x, breg[q].y);
            w3s[wave][r][c4 * 2 + 1] = pack2bf(breg[q].z, breg[q].w);
        }
        // issue next chunk's global loads (consumed after this chunk's WMMAs)
        const int dn = dk + 32;
        if (dn < DDIM) {
#pragma unroll
            for (int q = 0; q < 4; ++q) {
                int idx = q * 256 + tid;
                int r = idx >> 3, c4 = idx & 7;
                int s = slots[r];
                xreg[q] = make_float4(0.f, 0.f, 0.f, 0.f);
                if (s >= 0) xreg[q] = *(const float4*)(x + (size_t)(s >> 1) * DDIM + dn + c4 * 4);
            }
#pragma unroll
            for (int q = 0; q < 4; ++q) {
                int idx = q * 32 + lane;
                int r = idx >> 3, c4 = idx & 7;
                areg[q] = *(const float4*)(w1e + (size_t)(fbase + r) * DDIM + dn + c4 * 4);
                breg[q] = *(const float4*)(w3e + (size_t)(fbase + r) * DDIM + dn + c4 * 4);
            }
        }
        __syncthreads();   // staged chunk visible to all waves
        // B fragments once per wave, then sweep 8 row-subtiles
        Frag fb1, fb3;
        fb1.q[0] = *(const uint4*)&w1s[wave][ln][half * 4];
        fb1.q[1] = *(const uint4*)&w1s[wave][ln][8 + half * 4];
        fb3.q[0] = *(const uint4*)&w3s[wave][ln][half * 4];
        fb3.q[1] = *(const uint4*)&w3s[wave][ln][8 + half * 4];
#pragma unroll
        for (int sub = 0; sub < 8; ++sub) {
            Frag fa;
            fa.q[0] = *(const uint4*)&xs[sub * 16 + ln][half * 4];
            fa.q[1] = *(const uint4*)&xs[sub * 16 + ln][8 + half * 4];
            acch[sub] = __builtin_amdgcn_wmma_f32_16x16x32_bf16(false, fa.v, false, fb1.v,
                                                               (short)0, acch[sub], false, false);
            accg[sub] = __builtin_amdgcn_wmma_f32_16x16x32_bf16(false, fa.v, false, fb3.v,
                                                               (short)0, accg[sub], false, false);
        }
    }

    // SwiGLU (fast rcp) + store compact bf16 activations act[slot][F]
    const int fcol = fbase + ln;
#pragma unroll
    for (int sub = 0; sub < 8; ++sub) {
#pragma unroll
        for (int j = 0; j < 8; ++j) {
            int m = sub * 16 + j + half * 8;
            if (rowBlock * 128 + m < n_e) {
                int s = slots[m];
                float h = acch[sub][j], g = accg[sub][j];
                float sig = __builtin_amdgcn_rcpf(1.0f + __expf(-h));
                act[(size_t)s * FDIM + fcol] = f2bf(h * sig * g);
            }
        }
    }
}

// ------------------------------------------------ GEMM2 + weighted scatter
// Workgroup: one expert, 128 token rows x 128 D cols; same pipelined scheme.
__global__ void __launch_bounds__(256)
k_expert_gemm2(const unsigned short* __restrict__ act, const float* __restrict__ w2,
               const int* __restrict__ cnt, const int* __restrict__ list,
               const float* __restrict__ wslot, float* __restrict__ out) {
    const int e = blockIdx.y >> 4;
    const int rowBlock = blockIdx.y & 15;
    const int n_e = cnt[e];
    if (rowBlock * 128 >= n_e) return;
    const int dTile = blockIdx.x * 128;
    const int tid = threadIdx.x;
    const int wave = tid >> 5;
    const int lane = tid & 31;
    const int ln = lane & 15, half = lane >> 4;

    __shared__ unsigned int as_[128][16];       // 128 slots x 32 f bf16 pairs
    __shared__ unsigned int w2s[8][16][16];     // per-wave 16 d x 32 f
    __shared__ int slots[128];

    if (tid < 128) {
        int r = rowBlock * 128 + tid;
        slots[tid] = (r < n_e) ? list[e * T_TOK + r] : -1;
    }
    __syncthreads();

    const float* w2e = w2 + (size_t)e * DDIM * FDIM;
    const int dbase = dTile + wave * 16;

    uint4 areg[2];      // act already bf16: 128 rows x 4 uint4 = 512 / 256 thr
    float4 wreg[4];

    v8f acc[8];
#pragma unroll
    for (int s = 0; s < 8; ++s) acc[s] = {};

    // prologue loads, chunk 0
#pragma unroll
    for (int q = 0; q < 2; ++q) {
        int idx = q * 256 + tid;
        int r = idx >> 2, c = idx & 3;
        int s = slots[r];
        areg[q] = make_uint4(0u, 0u, 0u, 0u);
        if (s >= 0) areg[q] = *(const uint4*)(act + (size_t)s * FDIM + c * 8);
    }
#pragma unroll
    for (int q = 0; q < 4; ++q) {
        int idx = q * 32 + lane;
        int r = idx >> 3, c4 = idx & 7;
        wreg[q] = *(const float4*)(w2e + (size_t)(dbase + r) * FDIM + c4 * 4);
    }

    for (int fk = 0; fk < FDIM; fk += 32) {
        __syncthreads();
#pragma unroll
        for (int q = 0; q < 2; ++q) {
            int idx = q * 256 + tid;
            int r = idx >> 2, c = idx & 3;
            *(uint4*)&as_[r][c * 4] = areg[q];
        }
#pragma unroll
        for (int q = 0; q < 4; ++q) {
            int idx = q * 32 + lane;
            int r = idx >> 3, c4 = idx & 7;
            w2s[wave][r][c4 * 2 + 0] = pack2bf(wreg[q].x, wreg[q].y);
            w2s[wave][r][c4 * 2 + 1] = pack2bf(wreg[q].z, wreg[q].w);
        }
        const int fn = fk + 32;
        if (fn < FDIM) {
#pragma unroll
            for (int q = 0; q < 2; ++q) {
                int idx = q * 256 + tid;
                int r = idx >> 2, c = idx & 3;
                int s = slots[r];
                areg[q] = make_uint4(0u, 0u, 0u, 0u);
                if (s >= 0) areg[q] = *(const uint4*)(act + (size_t)s * FDIM + fn + c * 8);
            }
#pragma unroll
            for (int q = 0; q < 4; ++q) {
                int idx = q * 32 + lane;
                int r = idx >> 3, c4 = idx & 7;
                wreg[q] = *(const float4*)(w2e + (size_t)(dbase + r) * FDIM + fn + c4 * 4);
            }
        }
        __syncthreads();
        Frag fb;
        fb.q[0] = *(const uint4*)&w2s[wave][ln][half * 4];
        fb.q[1] = *(const uint4*)&w2s[wave][ln][8 + half * 4];
#pragma unroll
        for (int sub = 0; sub < 8; ++sub) {
            Frag fa;
            fa.q[0] = *(const uint4*)&as_[sub * 16 + ln][half * 4];
            fa.q[1] = *(const uint4*)&as_[sub * 16 + ln][8 + half * 4];
            acc[sub] = __builtin_amdgcn_wmma_f32_16x16x32_bf16(false, fa.v, false, fb.v,
                                                              (short)0, acc[sub], false, false);
        }
    }

    // weighted scatter: each out element gets exactly 2 atomic adds -> deterministic
    const int dcol = dbase + ln;
#pragma unroll
    for (int sub = 0; sub < 8; ++sub) {
#pragma unroll
        for (int j = 0; j < 8; ++j) {
            int m = sub * 16 + j + half * 8;
            if (rowBlock * 128 + m < n_e) {
                int s = slots[m];
                atomicAdd(&out[(size_t)(s >> 1) * DDIM + dcol], acc[sub][j] * wslot[s]);
            }
        }
    }
}

// ---------------------------------------------------------------- launcher
extern "C" void kernel_launch(void* const* d_in, const int* in_sizes, int n_in,
                              void* d_out, int out_size, void* d_ws, size_t ws_size,
                              hipStream_t stream) {
    (void)in_sizes; (void)n_in; (void)out_size; (void)ws_size;
    const float* x  = (const float*)d_in[0];   // [T, D]
    const float* gw = (const float*)d_in[1];   // [E, D]
    const float* w1 = (const float*)d_in[2];   // [E, F, D]
    const float* w2 = (const float*)d_in[3];   // [E, D, F]
    const float* w3 = (const float*)d_in[4];   // [E, F, D]
    float* out = (float*)d_out;                // [T, D]

    char* ws = (char*)d_ws;
    int*   cnt   = (int*)ws;                                   // E ints
    int*   list  = (int*)(ws + 256);                           // E*T ints
    float* wslot = (float*)(ws + 256 + NEXP * T_TOK * 4);      // 2T floats
    size_t actOff = ((256 + (size_t)NEXP * T_TOK * 4 + 2 * T_TOK * 4) + 255) & ~(size_t)255;
    unsigned short* act = (unsigned short*)(ws + actOff);      // 2T * F bf16 (~59 MB)

    k_zero<<<(T_TOK * DDIM + 255) / 256, 256, 0, stream>>>(out, T_TOK * DDIM, cnt);
    k_gate<<<T_TOK, 256, 0, stream>>>(x, gw, wslot, cnt, list);

    dim3 gA(FDIM / 128, NEXP * (T_TOK / 128));
    k_expert_gemm1<<<gA, 256, 0, stream>>>(x, w1, w3, cnt, list, act);

    dim3 gB(DDIM / 128, NEXP * (T_TOK / 128));
    k_expert_gemm2<<<gB, 256, 0, stream>>>(act, w2, cnt, list, wslot, out);
}
